// T2Net_3264175145105
// MI455X (gfx1250) — matrix operand order repaired and checked by reading, not compiled, not run on hardware
//
#include <hip/hip_runtime.h>
#include <math.h>

#define BN   4
#define NF   16
#define HH   64
#define WW   64
#define HW   4096
#define KPAD 160   // 144 real channels padded to 5 chunks of 32
#define KTB  4     // key tiles per stage (64 key rows, 21504B per LDS buffer)
#define NSTG (256 / KTB)
#define LDSS 168   // padded LDS row stride in halfs (336B -> bank-conflict-free)

typedef _Float16 half8  __attribute__((ext_vector_type(8)));
typedef _Float16 half16 __attribute__((ext_vector_type(16)));
typedef float    float8 __attribute__((ext_vector_type(8)));
typedef int      v4i    __attribute__((ext_vector_type(4)));

#if __has_builtin(__builtin_amdgcn_global_load_async_to_lds_b128) && \
    __has_builtin(__builtin_amdgcn_s_wait_asynccnt)
#define USE_ASYNC_LDS 1
typedef __attribute__((address_space(1))) v4i* gv4i_p;
typedef __attribute__((address_space(3))) v4i* lv4i_p;
#endif

// ---------------------------------------------------------------------------
// gradient magnitude: g = sqrt(gv^2 + gh^2 + 1e-6), zero-padded central diff
// ---------------------------------------------------------------------------
__global__ void __launch_bounds__(256)
k_grad(const float* __restrict__ t, float* __restrict__ g) {
  int idx = blockIdx.x * blockDim.x + threadIdx.x;
  if (idx >= BN * HW) return;
  int b = idx >> 12, p = idx & (HW - 1);
  int y = p >> 6, x = p & 63;
  const float* tb = t + (size_t)b * HW;
  float up = (y > 0)  ? tb[(y - 1) * WW + x] : 0.f;
  float dn = (y < 63) ? tb[(y + 1) * WW + x] : 0.f;
  float lf = (x > 0)  ? tb[y * WW + x - 1]   : 0.f;
  float rt = (x < 63) ? tb[y * WW + x + 1]   : 0.f;
  float v = dn - up, h = rt - lf;
  g[idx] = sqrtf(v * v + h * h + 1e-6f);
}

// ---------------------------------------------------------------------------
// generic direct conv KxK (zero pad K/2), optional residual add + relu
// ---------------------------------------------------------------------------
__global__ void __launch_bounds__(256)
k_conv(const float* __restrict__ in, const float* __restrict__ w,
       const float* __restrict__ bias, const float* __restrict__ res,
       float* __restrict__ out, int Cin, int Cout, int H, int W, int K,
       int doRelu, int Bn) {
  int idx = blockIdx.x * blockDim.x + threadIdx.x;
  int total = Bn * Cout * H * W;
  if (idx >= total) return;
  int x  = idx % W;
  int y  = (idx / W) % H;
  int co = (idx / (W * H)) % Cout;
  int b  = idx / (W * H * Cout);
  int pad = K / 2;
  const float* inb = in + (size_t)b * Cin * H * W;
  float acc = bias[co];
  for (int ci = 0; ci < Cin; ++ci) {
    const float* ic = inb + (size_t)ci * H * W;
    const float* wc = w + ((size_t)co * Cin + ci) * K * K;
    for (int i = 0; i < K; ++i) {
      int yy = y + i - pad;
      if (yy < 0 || yy >= H) continue;
      for (int j = 0; j < K; ++j) {
        int xx = x + j - pad;
        if (xx < 0 || xx >= W) continue;
        acc += wc[i * K + j] * ic[yy * W + xx];
      }
    }
  }
  if (res) acc += res[idx];
  if (doRelu) acc = fmaxf(acc, 0.f);
  out[idx] = acc;
}

// ---------------------------------------------------------------------------
// unfold(3x3,pad1) + column L2-normalize -> f16 rows [B][4096 pos][160 ch]
// channel index = c*9 + i*3 + j ; channels 144..159 zero padding
// ---------------------------------------------------------------------------
__global__ void __launch_bounds__(256)
k_unfold_norm(const float* __restrict__ x, _Float16* __restrict__ out) {
  int idx = blockIdx.x * blockDim.x + threadIdx.x;
  if (idx >= BN * HW) return;
  int b = idx >> 12, q = idx & (HW - 1);
  int y = q >> 6, x0 = q & 63;
  const float* xb = x + (size_t)b * NF * HW;
  float ss = 0.f;
  for (int c = 0; c < NF; ++c) {
    const float* xc = xb + c * HW;
    for (int i = 0; i < 3; ++i) {
      int yy = y + i - 1;
      if (yy < 0 || yy >= HH) continue;
      for (int j = 0; j < 3; ++j) {
        int xx = x0 + j - 1;
        if (xx < 0 || xx >= WW) continue;
        float v = xc[yy * WW + xx];
        ss += v * v;
      }
    }
  }
  float inv = 1.0f / fmaxf(sqrtf(ss), 1e-12f);
  _Float16* o = out + (size_t)idx * KPAD;
  for (int c = 0; c < NF; ++c) {
    const float* xc = xb + c * HW;
    for (int i = 0; i < 3; ++i)
      for (int j = 0; j < 3; ++j) {
        int yy = y + i - 1, xx = x0 + j - 1;
        float v = (yy >= 0 && yy < HH && xx >= 0 && xx < WW) ? xc[yy * WW + xx] : 0.f;
        o[c * 9 + i * 3 + j] = (_Float16)(v * inv);
      }
  }
  for (int k = 144; k < KPAD; ++k) o[k] = (_Float16)0.f;
}

// ---------------------------------------------------------------------------
// Fused correlation GEMM + max/argmax over keys.
// Block = 8 waves = 8 query tiles of one batch. Kn key rows double-buffered
// in LDS via async Global->LDS DMA: stage s+1 streams in (ASYNCcnt) while
// stage s is consumed by WMMA. Each thread issues exactly 5 async b128s per
// stage; ASYNCcnt completes in-order, so wait<=5 releases the previous stage.
// A-fragments batched as 10 ds_load_b128 then 5 WMMAs per key tile; B (query)
// chunks live in VGPRs throughout. max/argmax fused so R never materializes.
// ---------------------------------------------------------------------------
__global__ void __launch_bounds__(256)
k_corr_argmax(const _Float16* __restrict__ Kn, const _Float16* __restrict__ Qn,
              float* __restrict__ star, int* __restrict__ arg) {
  __shared__ __align__(32) _Float16 ldsK[2][KTB * 16 * LDSS];  // 2 x 21504 B

  const int tid    = threadIdx.x;
  const int lane   = tid & 31;
  const int waveId = tid >> 5;                 // 0..7
  const int wid    = blockIdx.x * 8 + waveId;  // 0..1023
  const int b  = wid >> 8;
  const int qt = wid & 255;
  const int n  = lane & 15;
  const bool hi = lane >= 16;

  // preload the 5 B-matrix chunks (invariant over key tiles): 40 VGPRs
  const _Float16* qrow = Qn + ((size_t)b * HW + qt * 16 + n) * KPAD + (hi ? 8 : 0);
  half16 Bm[5];
#pragma unroll
  for (int c = 0; c < 5; ++c) {
    half8 lo8 = *(const half8*)(qrow + c * 32);
    half8 hi8 = *(const half8*)(qrow + c * 32 + 16);
    Bm[c] = __builtin_shufflevector(lo8, hi8, 0,1,2,3,4,5,6,7,8,9,10,11,12,13,14,15);
  }

  const _Float16* kbase = Kn + (size_t)b * HW * KPAD;

  // stage copy: 64 key rows x 160 halfs = 1280 x 16B, exactly 5 per thread
  auto issueStage = [&](int stage, int buf) {
    const _Float16* gsrc = kbase + (size_t)stage * (KTB * 16) * KPAD;
    _Float16* lbase = &ldsK[buf][0];
#pragma unroll
    for (int it = 0; it < 5; ++it) {
      int t = tid + it * 256;
      int row = t / 20, seg = t - row * 20;
#if defined(USE_ASYNC_LDS)
      __builtin_amdgcn_global_load_async_to_lds_b128(
          (gv4i_p)(uintptr_t)(gsrc + (size_t)row * KPAD + seg * 8),
          (lv4i_p)(unsigned)(uintptr_t)(lbase + row * LDSS + seg * 8),
          0, 0);
#else
      *(half8*)(lbase + row * LDSS + seg * 8) =
          *(const half8*)(gsrc + (size_t)row * KPAD + seg * 8);
#endif
    }
  };

  float best = -3.4e38f;
  int bestIdx = 0;

  issueStage(0, 0);
  for (int s = 0; s < NSTG; ++s) {
    __syncthreads();  // all waves done reading buf[(s+1)&1] (stage s-1 data)
    if (s + 1 < NSTG) {
      issueStage(s + 1, (s + 1) & 1);   // prefetch next stage (overlaps WMMA)
#if defined(USE_ASYNC_LDS)
      __builtin_amdgcn_s_wait_asynccnt(5);  // stage s's 5 oldest completed
#endif
    } else {
#if defined(USE_ASYNC_LDS)
      __builtin_amdgcn_s_wait_asynccnt(0);
#endif
    }
    __syncthreads();  // stage s writes visible to all waves

    const _Float16* lb = &ldsK[s & 1][0];
#pragma unroll
    for (int kl = 0; kl < KTB; ++kl) {
      const _Float16* krow = lb + (kl * 16 + n) * LDSS + (hi ? 8 : 0);
      // batch all 5 chunk fragments (distinct regs -> clause + partial waits)
      union { half16 v; half8 h[2]; } A[5];
#pragma unroll
      for (int c = 0; c < 5; ++c) {
        A[c].h[0] = *(const half8*)(krow + c * 32);
        A[c].h[1] = *(const half8*)(krow + c * 32 + 16);
      }
      float8 acc = {};
#pragma unroll
      for (int c = 0; c < 5; ++c)
        acc = __builtin_amdgcn_wmma_f32_16x16x32_f16(
            false, A[c].v, false, Bm[c], (short)0, acc, false, false);
      int ktg = s * KTB + kl;
#pragma unroll
      for (int r = 0; r < 8; ++r) {
        float v = acc[r];
        int ki = ktg * 16 + r + (hi ? 8 : 0);
        if (v > best) { best = v; bestIdx = ki; }  // ascending -> first-max
      }
    }
  }
  // merge lane L with lane L^16 (key offsets 0-7 vs 8-15), tie -> lower index
  float ob = __shfl_xor(best, 16, 32);
  int   oi = __shfl_xor(bestIdx, 16, 32);
  if (ob > best || (ob == best && oi < bestIdx)) { best = ob; bestIdx = oi; }
  if (!hi) {
    star[b * HW + qt * 16 + n] = best;
    arg [b * HW + qt * 16 + n] = bestIdx;
  }
}

// ---------------------------------------------------------------------------
// hard-attention gather of V patches + fold(3x3,pad1)/9, without forming Vu
// ---------------------------------------------------------------------------
__global__ void __launch_bounds__(256)
k_gather_fold(const float* __restrict__ x2, const int* __restrict__ arg,
              float* __restrict__ T) {
  int idx = blockIdx.x * blockDim.x + threadIdx.x;
  if (idx >= BN * NF * HW) return;
  int p = idx & (HW - 1);
  int c = (idx >> 12) & (NF - 1);
  int b = idx >> 16;
  int y = p >> 6, x = p & 63;
  const float* xc = x2 + ((size_t)b * NF + c) * HW;
  const int* ab = arg + b * HW;
  float acc = 0.f;
  for (int i = 0; i < 3; ++i) {
    int qy = y + 1 - i;
    if (qy < 0 || qy >= HH) continue;
    for (int j = 0; j < 3; ++j) {
      int qx = x + 1 - j;
      if (qx < 0 || qx >= WW) continue;
      int a = ab[qy * WW + qx];
      int sy = (a >> 6) + i - 1, sx = (a & 63) + j - 1;
      if (sy >= 0 && sy < HH && sx >= 0 && sx < WW) acc += xc[sy * WW + sx];
    }
  }
  T[idx] = acc * (1.0f / 9.0f);
}

// ---------------------------------------------------------------------------
// 1x1 fuse conv on concat([x1, T]) scaled by star -> delta (added separately)
// ---------------------------------------------------------------------------
__global__ void __launch_bounds__(256)
k_fuse(const float* __restrict__ x1, const float* __restrict__ T,
       const float* __restrict__ w, const float* __restrict__ bias,
       const float* __restrict__ star, float* __restrict__ delta) {
  int idx = blockIdx.x * blockDim.x + threadIdx.x;
  if (idx >= BN * NF * HW) return;
  int p  = idx & (HW - 1);
  int co = (idx >> 12) & (NF - 1);
  int b  = idx >> 16;
  const float* x1b = x1 + (size_t)b * NF * HW;
  const float* Tb  = T  + (size_t)b * NF * HW;
  const float* wr  = w + (size_t)co * 2 * NF;
  float acc = bias[co];
  for (int ci = 0; ci < NF; ++ci) acc += wr[ci]      * x1b[ci * HW + p];
  for (int ci = 0; ci < NF; ++ci) acc += wr[NF + ci] * Tb [ci * HW + p];
  delta[idx] = acc * star[b * HW + p];
}

__global__ void __launch_bounds__(256)
k_add(float* __restrict__ out, const float* __restrict__ a,
      const float* __restrict__ c, int n) {
  int idx = blockIdx.x * blockDim.x + threadIdx.x;
  if (idx < n) out[idx] = a[idx] + c[idx];
}

// pixel shuffle r=2 : in [B, Cout*4, Ho/2, Wo/2] -> out [B, Cout, Ho, Wo]
__global__ void __launch_bounds__(256)
k_ps2(const float* __restrict__ in, float* __restrict__ out,
      int Cout, int Ho, int Wo, int Bn) {
  int idx = blockIdx.x * blockDim.x + threadIdx.x;
  int total = Bn * Cout * Ho * Wo;
  if (idx >= total) return;
  int x  = idx % Wo;
  int y  = (idx / Wo) % Ho;
  int co = (idx / (Wo * Ho)) % Cout;
  int b  = idx / (Wo * Ho * Cout);
  int ci = co * 4 + (y & 1) * 2 + (x & 1);
  out[idx] = in[(((size_t)b * (Cout * 4) + ci) * (Ho / 2) + (y >> 1)) * (Wo / 2) + (x >> 1)];
}

// ---------------------------------------------------------------------------
extern "C" void kernel_launch(void* const* d_in, const int* in_sizes, int n_in,
                              void* d_out, int out_size, void* d_ws, size_t ws_size,
                              hipStream_t stream) {
  (void)in_sizes; (void)n_in; (void)out_size; (void)ws_size;
  const float* t1 = (const float*)d_in[0];
  const float* t2 = (const float*)d_in[1];
  // params flattened in jax tree order (dict keys alphabetical, lists in order):
  // body1, body2, btail_b, btail_w, fuse, head1_b/w, head2_b/w, tail_b/w, up1_b/w, up2_b/w
  int pi = 2;
  const float *b1b1[4], *b1b2[4], *b1w1[4], *b1w2[4];
  const float *b2b1[4], *b2b2[4], *b2w1[4], *b2w2[4];
  for (int i = 0; i < 4; ++i) { b1b1[i]=(const float*)d_in[pi++]; b1b2[i]=(const float*)d_in[pi++];
                                b1w1[i]=(const float*)d_in[pi++]; b1w2[i]=(const float*)d_in[pi++]; }
  for (int i = 0; i < 4; ++i) { b2b1[i]=(const float*)d_in[pi++]; b2b2[i]=(const float*)d_in[pi++];
                                b2w1[i]=(const float*)d_in[pi++]; b2w2[i]=(const float*)d_in[pi++]; }
  const float* btail_b = (const float*)d_in[pi++];
  const float* btail_w = (const float*)d_in[pi++];
  const float *fb[4], *fw[4];
  for (int i = 0; i < 4; ++i) { fb[i]=(const float*)d_in[pi++]; fw[i]=(const float*)d_in[pi++]; }
  const float* head1_b = (const float*)d_in[pi++];
  const float* head1_w = (const float*)d_in[pi++];
  const float* head2_b = (const float*)d_in[pi++];
  const float* head2_w = (const float*)d_in[pi++];
  const float* tail_b  = (const float*)d_in[pi++];
  const float* tail_w  = (const float*)d_in[pi++];
  const float* up1_b   = (const float*)d_in[pi++];
  const float* up1_w   = (const float*)d_in[pi++];
  const float* up2_b   = (const float*)d_in[pi++];
  const float* up2_w   = (const float*)d_in[pi++];

  // workspace layout (float elements, 256B-aligned slices)
  float* Wf = (float*)d_ws;
  size_t off = 0;
  auto alloc = [&](size_t n) { float* p = Wf + off; off += (n + 63) & ~(size_t)63; return p; };
  const size_t nG = (size_t)BN * HW;          // 16384
  const size_t nX = (size_t)BN * NF * HW;     // 262144
  float* g1   = alloc(nG);
  float* g2   = alloc(nG);
  float* x1   = alloc(nX);
  float* x2   = alloc(nX);
  float* res1 = alloc(nX);
  float* res2 = alloc(nX);
  float* tmp  = alloc(nX);
  float* Tb   = alloc(nX);
  float* star = alloc(nG);
  int*   argB = (int*)alloc(nG);
  _Float16* Qn = (_Float16*)alloc((size_t)BN * HW * KPAD / 2);
  _Float16* Kn = (_Float16*)alloc((size_t)BN * HW * KPAD / 2);
  float* ub1 = alloc((size_t)BN * 64 * HW);        // 64x64 conv out, 64ch
  float* us1 = alloc((size_t)BN * NF * HW * 4);    // 128x128, 16ch
  float* ub2 = alloc((size_t)BN * 64 * HW * 4);    // 128x128, 64ch
  float* us2 = alloc((size_t)BN * NF * HW * 16);   // 256x256, 16ch

  auto grid = [](size_t n) { return (unsigned)((n + 255) / 256); };

  // forward swaps: t1,t2 = t2,t1  => branch1 from input t2, branch2 from t1
  k_grad<<<grid(nG), 256, 0, stream>>>(t2, g1);
  k_grad<<<grid(nG), 256, 0, stream>>>(t1, g2);
  k_conv<<<grid(nX), 256, 0, stream>>>(g1, head1_w, head1_b, nullptr, x1, 1, NF, HH, WW, 5, 0, BN);
  k_conv<<<grid(nX), 256, 0, stream>>>(g2, head2_w, head2_b, nullptr, x2, 1, NF, HH, WW, 5, 0, BN);
  (void)hipMemcpyAsync(res1, x1, nX * sizeof(float), hipMemcpyDeviceToDevice, stream);
  (void)hipMemcpyAsync(res2, x2, nX * sizeof(float), hipMemcpyDeviceToDevice, stream);

  for (int i = 0; i < 4; ++i) {
    // resblocks (in-place residual add is per-index safe)
    k_conv<<<grid(nX), 256, 0, stream>>>(x1, b1w1[i], b1b1[i], nullptr, tmp, NF, NF, HH, WW, 3, 1, BN);
    k_conv<<<grid(nX), 256, 0, stream>>>(tmp, b1w2[i], b1b2[i], x1, x1, NF, NF, HH, WW, 3, 0, BN);
    k_conv<<<grid(nX), 256, 0, stream>>>(x2, b2w1[i], b2b1[i], nullptr, tmp, NF, NF, HH, WW, 3, 1, BN);
    k_conv<<<grid(nX), 256, 0, stream>>>(tmp, b2w2[i], b2b2[i], x2, x2, NF, NF, HH, WW, 3, 0, BN);
    // transformer: Q from x1, K=V from x2
    k_unfold_norm<<<grid(nG), 256, 0, stream>>>(x1, Qn);
    k_unfold_norm<<<grid(nG), 256, 0, stream>>>(x2, Kn);
    k_corr_argmax<<<128, 256, 0, stream>>>(Kn, Qn, star, argB);
    k_gather_fold<<<grid(nX), 256, 0, stream>>>(x2, argB, Tb);
    k_fuse<<<grid(nX), 256, 0, stream>>>(x1, Tb, fw[i], fb[i], star, tmp);
    k_add<<<grid(nX), 256, 0, stream>>>(x1, x1, tmp, (int)nX);
  }

  // upsample branch (y1)
  k_add<<<grid(nX), 256, 0, stream>>>(tmp, x1, res1, (int)nX);
  size_t nU1 = (size_t)BN * 64 * HW;
  k_conv<<<grid(nU1), 256, 0, stream>>>(tmp, up1_w, up1_b, nullptr, ub1, NF, 64, HH, WW, 3, 0, BN);
  size_t nS1 = (size_t)BN * NF * HW * 4;
  k_ps2<<<grid(nS1), 256, 0, stream>>>(ub1, us1, NF, 128, 128, BN);
  size_t nU2 = (size_t)BN * 64 * HW * 4;
  k_conv<<<grid(nU2), 256, 0, stream>>>(us1, up2_w, up2_b, nullptr, ub2, NF, 64, 128, 128, 3, 0, BN);
  size_t nS2 = (size_t)BN * NF * HW * 16;
  k_ps2<<<grid(nS2), 256, 0, stream>>>(ub2, us2, NF, 256, 256, BN);
  float* y2 = (float*)d_out;                 // [4,1,64,64]  first in tuple
  float* y1 = (float*)d_out + BN * HW;       // [4,1,256,256]
  size_t nY1 = (size_t)BN * 256 * 256;
  k_conv<<<grid(nY1), 256, 0, stream>>>(us2, tail_w, tail_b, nullptr, y1, NF, 1, 256, 256, 3, 0, BN);

  // y2 branch
  k_add<<<grid(nX), 256, 0, stream>>>(tmp, x2, res2, (int)nX);
  k_conv<<<grid(nG), 256, 0, stream>>>(tmp, btail_w, btail_b, nullptr, y2, NF, 1, HH, WW, 1, 0, BN);
}
